// SparseHigherOrderTensorDynamics_88304527606108
// MI455X (gfx1250) — matrix operand order repaired and checked by reading, not compiled, not run on hardware
//
#include <hip/hip_runtime.h>
#include <hip/hip_bf16.h>

typedef __attribute__((ext_vector_type(16))) _Float16 v16h;
typedef __attribute__((ext_vector_type(4)))  _Float16 v4h;
typedef __attribute__((ext_vector_type(8)))  float    v8f;

#define BDIM 32768
#define DDIM 1024
#define TOPK 102

// Async global->LDS copy via inline asm (CDNA5 GLOBAL_LOAD_ASYNC_TO_LDS_B128).
// Flip to 0 to fall back to synchronous b128 load/store staging.
#define USE_ASYNC_ASM 1

// ---------------- mask pipeline ----------------

__global__ void zero_kernel(float* p, int n) {
    int i = blockIdx.x * blockDim.x + threadIdx.x;
    if (i < n) p[i] = 0.0f;
}

// Per-column sum of |x|; block b handles rows [b*128, b*128+128).
__global__ __launch_bounds__(256) void colabs_kernel(const float* __restrict__ x,
                                                     float* __restrict__ colsum) {
    const int t = threadIdx.x;
    const int r0 = blockIdx.x * 128;
    float a0 = 0.f, a1 = 0.f, a2 = 0.f, a3 = 0.f;
    for (int r = 0; r < 128; ++r) {
        const float* row = x + (size_t)(r0 + r) * DDIM;
        a0 += fabsf(row[t]);
        a1 += fabsf(row[t + 256]);
        a2 += fabsf(row[t + 512]);
        a3 += fabsf(row[t + 768]);
    }
    atomicAdd(&colsum[t],       a0);
    atomicAdd(&colsum[t + 256], a1);
    atomicAdd(&colsum[t + 512], a2);
    atomicAdd(&colsum[t + 768], a3);
}

// Top-k mask via rank selection; stable tie-break on lower index (jax top_k order).
__global__ __launch_bounds__(1024) void build_mask_kernel(const float* __restrict__ colsum,
                                                          float* __restrict__ maskv) {
    __shared__ float v[DDIM];
    const int t = threadIdx.x;
    v[t] = colsum[t];
    __syncthreads();
    const float mv = v[t];
    int rank = 0;
    for (int j = 0; j < DDIM; ++j) {
        float vj = v[j];
        rank += (vj > mv) || (vj == mv && j < t);
    }
    maskv[t] = (rank < TOPK) ? 1.0f : 0.0f;
}

// ---------------- precision conversion ----------------

__global__ __launch_bounds__(256) void convert_x_kernel(const float* __restrict__ x,
                                                        _Float16* __restrict__ xh) {
    size_t i = (size_t)(blockIdx.x * blockDim.x + threadIdx.x) * 4;
    float4 v = *(const float4*)(x + i);
    v4h h = { (_Float16)v.x, (_Float16)v.y, (_Float16)v.z, (_Float16)v.w };
    *(v4h*)(xh + i) = h;
}

// w[k][n] -> whT[n][k] in f16, optionally row-masked (fold x_sparse's mask into W:
// (x*mask)@w == x@(diag(mask)*w)).  Tiled 32x32 transpose through LDS.
template <int MASKED>
__global__ __launch_bounds__(256) void convert_wT_kernel(const float* __restrict__ w,
                                                         const float* __restrict__ maskv,
                                                         _Float16* __restrict__ whT) {
    __shared__ _Float16 tile[32][33];
    const int k0 = blockIdx.x * 32;
    const int n0 = blockIdx.y * 32;
    const int tx = threadIdx.x & 31;
    const int ty = threadIdx.x >> 5;  // 0..7
#pragma unroll
    for (int j = 0; j < 4; ++j) {
        int k = ty + 8 * j;
        float v = w[(size_t)(k0 + k) * DDIM + (n0 + tx)];
        if (MASKED) v *= maskv[k0 + k];
        tile[k][tx] = (_Float16)v;
    }
    __syncthreads();
#pragma unroll
    for (int j = 0; j < 4; ++j) {
        int n = ty + 8 * j;
        whT[(size_t)(n0 + n) * DDIM + (k0 + tx)] = tile[tx][n];
    }
}

// ---------------- WMMA GEMM + fused epilogue ----------------
// MODE 0: out = acc                       (w unmasked)
// MODE 1: out = acc * x[b,n]*mask[n]      (w row-masked at conversion)
// MODE 2: out = acc^2 * x[b,n]*mask[n]
//
// Block 256 threads = 8 waves (4M x 2N).  Block tile 128x128, K-step 32.
// Wave tile 32x64 -> 8 v_wmma_f32_16x16x32_f16 per K-step.
// Double-buffered LDS; next tile staged via async global->LDS copies that
// overlap with the current tile's WMMA work.
template <int MODE>
__global__ __launch_bounds__(256) void gemm_wmma_kernel(const _Float16* __restrict__ Xh,
                                                        const _Float16* __restrict__ WhT,
                                                        const float* __restrict__ X,
                                                        const float* __restrict__ maskv,
                                                        float* __restrict__ out) {
    // pitch 40 halves = 80B (16B aligned) -> b128 chunks stay aligned
    __shared__ _Float16 sA[2][128][40] __attribute__((aligned(16)));   // A (M x K)
    __shared__ _Float16 sBt[2][128][40] __attribute__((aligned(16)));  // B col-major (N x K)

    const int tid  = threadIdx.x;
    const int lane = tid & 31;
    const int wid  = tid >> 5;
    const int wm   = wid >> 1;     // 0..3
    const int wn   = wid & 1;      // 0..1
    const int l15  = lane & 15;
    const int hi   = (lane >= 16);
    const int kbA  = hi ? 8  : 0;  // ISA 16-bit A 16x32 layout
    const int kbB  = hi ? 8  : 0;  // B: same K layout as A

    const int m0 = blockIdx.y * 128;
    const int n0 = blockIdx.x * 128;

    // ---- staging: 512 16B-chunks per tile, 2/thread; addresses loop-invariant ----
    const int rA0 = tid >> 2,          cA0 = (tid & 3) * 8;
    const int rA1 = (tid + 256) >> 2,  cA1 = ((tid + 256) & 3) * 8;
    const _Float16* gA0 = Xh  + (size_t)(m0 + rA0) * DDIM + cA0;
    const _Float16* gA1 = Xh  + (size_t)(m0 + rA1) * DDIM + cA1;
    const _Float16* gB0 = WhT + (size_t)(n0 + rA0) * DDIM + cA0;
    const _Float16* gB1 = WhT + (size_t)(n0 + rA1) * DDIM + cA1;
    // LDS byte offsets (low 32 bits of generic LDS pointer = wave LDS offset)
    uint32_t lA0[2], lA1[2], lB0[2], lB1[2];
#pragma unroll
    for (int b = 0; b < 2; ++b) {
        lA0[b] = (uint32_t)(uintptr_t)&sA[b][rA0][cA0];
        lA1[b] = (uint32_t)(uintptr_t)&sA[b][rA1][cA1];
        lB0[b] = (uint32_t)(uintptr_t)&sBt[b][rA0][cA0];
        lB1[b] = (uint32_t)(uintptr_t)&sBt[b][rA1][cA1];
    }

    // loop-invariant fragment pointers
    const _Float16* ap0[2] = { &sA[0][wm * 32 + l15][0],      &sA[1][wm * 32 + l15][0] };
    const _Float16* ap1[2] = { &sA[0][wm * 32 + 16 + l15][0], &sA[1][wm * 32 + 16 + l15][0] };
    const _Float16* bp[2][4];
#pragma unroll
    for (int b = 0; b < 2; ++b)
#pragma unroll
        for (int t = 0; t < 4; ++t) bp[b][t] = &sBt[b][wn * 64 + t * 16 + l15][0];

#if USE_ASYNC_ASM
#define STAGE_TILE(buf, koff)                                                          \
    do {                                                                               \
        asm volatile("global_load_async_to_lds_b128 %0, %1, off"                       \
                     :: "v"(lA0[buf]), "v"((uint64_t)(uintptr_t)(gA0 + (koff)))        \
                     : "memory");                                                      \
        asm volatile("global_load_async_to_lds_b128 %0, %1, off"                       \
                     :: "v"(lA1[buf]), "v"((uint64_t)(uintptr_t)(gA1 + (koff)))        \
                     : "memory");                                                      \
        asm volatile("global_load_async_to_lds_b128 %0, %1, off"                       \
                     :: "v"(lB0[buf]), "v"((uint64_t)(uintptr_t)(gB0 + (koff)))        \
                     : "memory");                                                      \
        asm volatile("global_load_async_to_lds_b128 %0, %1, off"                       \
                     :: "v"(lB1[buf]), "v"((uint64_t)(uintptr_t)(gB1 + (koff)))        \
                     : "memory");                                                      \
    } while (0)
#define WAIT_ASYNC_LE4() asm volatile("s_wait_asynccnt 0x4" ::: "memory")
#define WAIT_ASYNC_0()   asm volatile("s_wait_asynccnt 0x0" ::: "memory")
#else
#define STAGE_TILE(buf, koff)                                                          \
    do {                                                                               \
        *(uint4*)(uintptr_t)(size_t)lA0[buf] = *(const uint4*)(gA0 + (koff));          \
        *(uint4*)(uintptr_t)(size_t)lA1[buf] = *(const uint4*)(gA1 + (koff));          \
        *(uint4*)(uintptr_t)(size_t)lB0[buf] = *(const uint4*)(gB0 + (koff));          \
        *(uint4*)(uintptr_t)(size_t)lB1[buf] = *(const uint4*)(gB1 + (koff));          \
    } while (0)
#define WAIT_ASYNC_LE4()
#define WAIT_ASYNC_0()
#endif

    v8f acc[2][4];
#pragma unroll
    for (int m = 0; m < 2; ++m)
#pragma unroll
        for (int t = 0; t < 4; ++t)
            acc[m][t] = (v8f){0.f, 0.f, 0.f, 0.f, 0.f, 0.f, 0.f, 0.f};

#if !USE_ASYNC_ASM
    // sync fallback uses generic-pointer LDS stores; recompute pointers
#pragma unroll
    for (int b = 0; b < 2; ++b) {
        lA0[b] = 0; lA1[b] = 0; lB0[b] = 0; lB1[b] = 0;  // unused
    }
#endif

#if USE_ASYNC_ASM
    STAGE_TILE(0, 0);  // prologue: tile 0 in flight
#endif

#pragma unroll 2
    for (int i = 0; i < DDIM / 32; ++i) {
        const int cur = i & 1;
        __syncthreads();  // all waves done reading buf[cur^1] (iteration i-1)
#if USE_ASYNC_ASM
        if (i + 1 < DDIM / 32) {
            STAGE_TILE(cur ^ 1, (i + 1) * 32);  // overlap next tile with compute
            WAIT_ASYNC_LE4();                   // older 4 (buf[cur]) have landed
        } else {
            WAIT_ASYNC_0();
        }
#else
        {
            const int koff = i * 32;
            *(uint4*)&sA[cur][rA0][cA0]  = *(const uint4*)(gA0 + koff);
            *(uint4*)&sA[cur][rA1][cA1]  = *(const uint4*)(gA1 + koff);
            *(uint4*)&sBt[cur][rA0][cA0] = *(const uint4*)(gB0 + koff);
            *(uint4*)&sBt[cur][rA1][cA1] = *(const uint4*)(gB1 + koff);
        }
#endif
        __syncthreads();  // staged tile visible to all waves

        v16h a0, a1;
#pragma unroll
        for (int e = 0; e < 8; ++e) {
            a0[e]     = ap0[cur][kbA + e];
            a0[8 + e] = ap0[cur][kbA + 16 + e];
            a1[e]     = ap1[cur][kbA + e];
            a1[8 + e] = ap1[cur][kbA + 16 + e];
        }

#pragma unroll
        for (int t = 0; t < 4; ++t) {
            v16h b;
#pragma unroll
            for (int e = 0; e < 16; ++e) b[e] = bp[cur][t][kbB + (e & 7) + (e >> 3) * 16];
            acc[0][t] = __builtin_amdgcn_wmma_f32_16x16x32_f16(
                false, a0, false, b, (short)0, acc[0][t], false, false);
            acc[1][t] = __builtin_amdgcn_wmma_f32_16x16x32_f16(
                false, a1, false, b, (short)0, acc[1][t], false, false);
        }
    }

    // ---- epilogue: C layout VGPR r -> M = r (lanes 0-15) / 8+r (lanes 16-31) ----
#pragma unroll
    for (int m = 0; m < 2; ++m) {
        const int rbase = m0 + wm * 32 + m * 16 + (hi ? 8 : 0);
#pragma unroll
        for (int t = 0; t < 4; ++t) {
            const int col  = n0 + wn * 64 + t * 16 + l15;
            const float mc = (MODE != 0) ? maskv[col] : 1.0f;
#pragma unroll
            for (int r = 0; r < 8; ++r) {
                const int row = rbase + r;
                float v = acc[m][t][r];
                if (MODE == 1) {
                    v *= X[(size_t)row * DDIM + col] * mc;
                } else if (MODE == 2) {
                    v = v * v * (X[(size_t)row * DDIM + col] * mc);
                }
                out[(size_t)row * DDIM + col] = v;
            }
        }
    }
#undef STAGE_TILE
#undef WAIT_ASYNC_LE4
#undef WAIT_ASYNC_0
}

// ---------------- launch ----------------

extern "C" void kernel_launch(void* const* d_in, const int* in_sizes, int n_in,
                              void* d_out, int out_size, void* d_ws, size_t ws_size,
                              hipStream_t stream) {
    (void)in_sizes; (void)n_in; (void)out_size; (void)ws_size;
    const float* x  = (const float*)d_in[0];
    const float* w1 = (const float*)d_in[1];
    const float* w2 = (const float*)d_in[2];
    const float* w3 = (const float*)d_in[3];
    float* out = (float*)d_out;

    char* ws = (char*)d_ws;
    float*    colsum = (float*)ws;                   // 4 KB
    float*    maskv  = (float*)(ws + 4096);          // 4 KB
    _Float16* whT1   = (_Float16*)(ws + 8192);       // 2 MB each
    _Float16* whT2   = whT1 + (1 << 20);
    _Float16* whT3   = whT2 + (1 << 20);
    _Float16* xh     = whT3 + (1 << 20);             // 64 MB

    // 1) mask pipeline
    zero_kernel<<<4, 256, 0, stream>>>(colsum, DDIM);
    colabs_kernel<<<BDIM / 128, 256, 0, stream>>>(x, colsum);
    build_mask_kernel<<<1, 1024, 0, stream>>>(colsum, maskv);

    // 2) conversions: x -> f16; w -> f16 transposed (w2/w3 row-masked)
    convert_x_kernel<<<(BDIM * (DDIM / 4)) / 256, 256, 0, stream>>>(x, xh);
    dim3 tgrid(DDIM / 32, DDIM / 32);
    convert_wT_kernel<0><<<tgrid, 256, 0, stream>>>(w1, maskv, whT1);
    convert_wT_kernel<1><<<tgrid, 256, 0, stream>>>(w2, maskv, whT2);
    convert_wT_kernel<1><<<tgrid, 256, 0, stream>>>(w3, maskv, whT3);

    // 3) three WMMA GEMM passes with fused epilogues
    dim3 grid(DDIM / 128, BDIM / 128);  // (8, 256)
    dim3 blk(256);
    const size_t plane = (size_t)BDIM * DDIM;
    gemm_wmma_kernel<0><<<grid, blk, 0, stream>>>(xh, whT1, x, maskv, out);
    gemm_wmma_kernel<1><<<grid, blk, 0, stream>>>(xh, whT2, x, maskv, out + plane);
    gemm_wmma_kernel<2><<<grid, blk, 0, stream>>>(xh, whT3, x, maskv, out + 2 * plane);
}
